// BERTEncoder_6459630813513
// MI455X (gfx1250) — matrix operand order repaired and checked by reading, not compiled, not run on hardware
//
#include <hip/hip_runtime.h>
#include <hip/hip_bf16.h>
#include <cstddef>
#include <cstdint>

// ---------------- model constants ----------------
#define BB   16
#define SS   512
#define DD   768
#define HH   12
#define LL   12
#define FF   3072
#define DH   64
#define MROWS (BB*SS)          // 8192
#define SCALE_QK 0.125f        // 1/sqrt(64)

// ---------------- WMMA types ----------------
typedef __attribute__((ext_vector_type(16))) __bf16 v16bf;
typedef __attribute__((ext_vector_type(8)))  __bf16 v8bf;
typedef __attribute__((ext_vector_type(8)))  float  v8f;

union AFrag { v16bf v; v8bf h[2]; };

// ---------------- tiling ----------------
#define BM 128
#define BK 32
#define LDA_S 40   // bf16 elems per LDS row (80B rows, every 16B chunk aligned)
#define LDB_S 40

// =====================================================================
// Embedding: X = tok_emb[tok] + seg_emb[seg] + pos_emb[:S]  (fp32 + bf16)
// =====================================================================
__global__ __launch_bounds__(256)
void embed_kernel(const int* __restrict__ tokens, const int* __restrict__ segments,
                  const float* __restrict__ tok_emb, const float* __restrict__ seg_emb,
                  const float* __restrict__ pos_emb,
                  float* __restrict__ X, __bf16* __restrict__ Xh) {
    size_t idx = (size_t)blockIdx.x * 256 + threadIdx.x;
    if (idx >= (size_t)MROWS * DD) return;
    int d  = (int)(idx % DD);
    int bs = (int)(idx / DD);
    int s  = bs % SS;
    float v = tok_emb[(size_t)tokens[bs] * DD + d]
            + seg_emb[(size_t)segments[bs] * DD + d]
            + pos_emb[(size_t)s * DD + d];
    X[idx]  = v;
    Xh[idx] = (__bf16)v;
}

// =====================================================================
// Weight convert + transpose: W[K,N] fp32 -> Wt[N,K] bf16
// =====================================================================
__global__ __launch_bounds__(256)
void w_to_bf16t(const float* __restrict__ W, __bf16* __restrict__ Wt, int Kd, int Nd) {
    size_t idx = (size_t)blockIdx.x * 256 + threadIdx.x;
    if (idx >= (size_t)Kd * Nd) return;
    int k = (int)(idx / Nd);
    int n = (int)(idx % Nd);
    Wt[(size_t)n * Kd + k] = (__bf16)W[idx];
}

// =====================================================================
// WMMA GEMM, bf16 operands, double-buffered async-to-LDS staging.
//  A:  bf16, rows (M) x K, row stride aStride  (K contiguous)
//  Bt: bf16, rows (N) x K, row stride btStride (K contiguous)
//  MODE 0: X@W: fp32 C (+bias,+resid,+relu), optional bf16 copy
//          (cbmode 1: [M,ldc] layout; cbmode 2: V-transpose [b,h,dh,s])
//  MODE 1: QK^T -> fp32 P * scale               (z = b*H+h)
//  MODE 2: P@V  -> bf16 Oh at [b*S+m, h*64+n]
//  Tiling: MODE 0/1: block 128x128, wave 64x32 (8 wmma / k-step)
//          MODE 2:   block 128x64,  wave 32x32 (4 wmma / k-step)
// =====================================================================
template<int MODE>
__global__ __launch_bounds__(256)
void gemm_wmma(const __bf16* __restrict__ A, const __bf16* __restrict__ Bt,
               const float* __restrict__ bias, const float* __restrict__ resid,
               float* __restrict__ C, __bf16* __restrict__ Cb,
               int K, int ldc, int aStride, int btStride,
               int relu, int cbmode, float scale) {
    constexpr int BN_ = (MODE == 2) ? 64 : 128;   // block N
    constexpr int MI  = (MODE == 2) ? 2 : 4;      // 16-row subtiles per wave
    constexpr int NI  = 2;                        // 16-col subtiles per wave
    constexpr int WMW = (MODE == 2) ? 4 : 2;      // waves along M
    constexpr int BTRIP = (BN_ * 4) / 256;        // B staging trips (1 or 2)

    __shared__ __bf16 ldsA[2][BM  * LDA_S];
    __shared__ __bf16 ldsB[2][BN_ * LDB_S];

    const int tid  = threadIdx.x;
    const int lane = tid & 31;
    const int wave = tid >> 5;
    const int rm0  = (wave % WMW) * (MI * 16);
    const int cn0  = (wave / WMW) * (NI * 16);

    const int bx = blockIdx.x, by = blockIdx.y, z = blockIdx.z;
    const int bh_b = z / HH, bh_h = z % HH;
    const int bn0 = bx * BN_;

    // per-mode base pointers (both operands contiguous along K)
    const __bf16* aBase; const __bf16* btBase;
    if (MODE == 0) {
        aBase  = A  + (size_t)by * BM * aStride;
        btBase = Bt + (size_t)bn0 * btStride;
    } else if (MODE == 1) {
        aBase  = A  + ((size_t)(bh_b*SS + by*BM)) * DD + bh_h*DH;
        btBase = Bt + ((size_t)(bh_b*SS + bn0))   * DD + bh_h*DH;
    } else {
        aBase  = A  + ((size_t)z * SS + by*BM) * SS;
        btBase = Bt + ((size_t)z * DH + bn0)   * SS;
    }

    // async DMA stage of one (A,B) k-tile into LDS buffer `buf`
    auto stage = [&](int buf, int kb) {
#pragma unroll
        for (int it = 0; it < 2; ++it) {             // A: 128 rows x 4 x 16B
            int i = tid + it * 256;
            int row = i >> 2, seg = i & 3;
            unsigned long long g = (unsigned long long)(uintptr_t)
                (aBase + (size_t)row * aStride + kb + seg * 8);
            unsigned l = (unsigned)(uintptr_t)(&ldsA[buf][row * LDA_S + seg * 8]);
            asm volatile("global_load_async_to_lds_b128 %0, %1, off"
                         :: "v"(l), "v"(g) : "memory");
        }
#pragma unroll
        for (int it = 0; it < BTRIP; ++it) {         // B: BN_ rows x 4 x 16B
            int i = tid + it * 256;
            int row = i >> 2, seg = i & 3;
            unsigned long long g = (unsigned long long)(uintptr_t)
                (btBase + (size_t)row * btStride + kb + seg * 8);
            unsigned l = (unsigned)(uintptr_t)(&ldsB[buf][row * LDB_S + seg * 8]);
            asm volatile("global_load_async_to_lds_b128 %0, %1, off"
                         :: "v"(l), "v"(g) : "memory");
        }
    };

    // fragment lane layout (wave32)
    const int lan15 = lane & 15;
    const int klo   = (lane < 16) ? 0 : 8;    // A: j<8 -> K=klo+j ; j>=8 -> K=16+klo+(j-8)
    const int kboff = (lane < 16) ? 0 : 16;   // B: same K order as A (loaded at klo); kboff unused

    v8f acc[MI][NI];
#pragma unroll
    for (int i = 0; i < MI; ++i)
#pragma unroll
        for (int j = 0; j < NI; ++j) acc[i][j] = {};

    const int nk = K / BK;
    stage(0, 0);
    for (int k = 0; k < nk; ++k) {
        if (k + 1 < nk) {
            stage((k + 1) & 1, (k + 1) * BK);
            // in-order completion: wait until only next tile's copies remain
            if (MODE == 2) asm volatile("s_wait_asynccnt 3" ::: "memory");
            else           asm volatile("s_wait_asynccnt 4" ::: "memory");
        } else {
            asm volatile("s_wait_asynccnt 0" ::: "memory");
        }
        __syncthreads();

        const __bf16* bufA = ldsA[k & 1];
        const __bf16* bufB = ldsB[k & 1];

        AFrag afr[MI], bfr[NI];
#pragma unroll
        for (int i = 0; i < MI; ++i) {
            const __bf16* pr = &bufA[(rm0 + i*16 + lan15) * LDA_S];
            afr[i].h[0] = *(const v8bf*)(pr + klo);
            afr[i].h[1] = *(const v8bf*)(pr + 16 + klo);
        }
#pragma unroll
        for (int j = 0; j < NI; ++j) {
            const __bf16* pc = &bufB[(cn0 + j*16 + lan15) * LDB_S];
            bfr[j].h[0] = *(const v8bf*)(pc + klo);
            bfr[j].h[1] = *(const v8bf*)(pc + 16 + klo);
        }

#pragma unroll
        for (int i = 0; i < MI; ++i)
#pragma unroll
            for (int j = 0; j < NI; ++j)
                acc[i][j] = __builtin_amdgcn_wmma_f32_16x16x32_bf16(
                    false, afr[i].v, false, bfr[j].v,
                    (short)0, acc[i][j], false, false);

        __syncthreads();
    }

    // ---- epilogue: element r -> M = r + (lane<16?0:8), N = lane&15 ----
    const int half = (lane < 16) ? 0 : 8;
#pragma unroll
    for (int i = 0; i < MI; ++i) {
#pragma unroll
        for (int j = 0; j < NI; ++j) {
#pragma unroll
            for (int r = 0; r < 8; ++r) {
                int gm = by * BM + rm0 + i*16 + half + r;
                int gn = bn0 + cn0 + j*16 + lan15;
                float v = acc[i][j][r];
                if (MODE == 0) {
                    v += bias[gn];
                    if (resid) v += resid[(size_t)gm * ldc + gn];
                    if (relu)  v = fmaxf(v, 0.0f);
                    if (C) C[(size_t)gm * ldc + gn] = v;
                    if (cbmode == 1) {
                        Cb[(size_t)gm * ldc + gn] = (__bf16)v;
                    } else if (cbmode == 2) {
                        int b2 = gm >> 9, s2 = gm & 511;
                        int h2 = gn >> 6, d2 = gn & 63;
                        Cb[((size_t)(b2*HH + h2) * DH + d2) * SS + s2] = (__bf16)v;
                    }
                } else if (MODE == 1) {
                    C[((size_t)z * SS + gm) * SS + gn] = v * scale;
                } else {
                    Cb[((size_t)(bh_b*SS + gm)) * DD + bh_h*DH + gn] = (__bf16)v;
                }
            }
        }
    }
}

// =====================================================================
// Masked softmax: reads fp32 scores P, writes bf16 probs Ph.
// mask: key >= valid_len -> -1e6 (matches reference)
// =====================================================================
__global__ __launch_bounds__(256)
void softmax_mask_kernel(const float* __restrict__ P, const int* __restrict__ valid_lens,
                         __bf16* __restrict__ Ph) {
    __shared__ float red[256];
    const int row = blockIdx.x;
    const int b   = row / (HH * SS);
    const int vl  = valid_lens[b];
    const float* p = P + (size_t)row * SS;
    __bf16* ph = Ph + (size_t)row * SS;
    const int t = threadIdx.x;

    float v0 = (t        < vl) ? p[t]       : -1e6f;
    float v1 = ((t+256)  < vl) ? p[t + 256] : -1e6f;

    red[t] = fmaxf(v0, v1);
    __syncthreads();
    for (int s = 128; s > 0; s >>= 1) { if (t < s) red[t] = fmaxf(red[t], red[t+s]); __syncthreads(); }
    float m = red[0];
    __syncthreads();

    float e0 = __expf(v0 - m), e1 = __expf(v1 - m);
    red[t] = e0 + e1;
    __syncthreads();
    for (int s = 128; s > 0; s >>= 1) { if (t < s) red[t] += red[t+s]; __syncthreads(); }
    float inv = 1.0f / red[0];

    ph[t]       = (__bf16)(e0 * inv);
    ph[t + 256] = (__bf16)(e1 * inv);
}

// =====================================================================
// LayerNorm over D=768, one block per row; fp32 out + optional bf16 out
// =====================================================================
__global__ __launch_bounds__(256)
void layernorm_kernel(const float* __restrict__ Z, const float* __restrict__ g,
                      const float* __restrict__ beta, float* __restrict__ out,
                      __bf16* __restrict__ outh) {
    __shared__ float red[256];
    const int row = blockIdx.x;
    const int t = threadIdx.x;
    const float* zp = Z + (size_t)row * DD;

    float x0 = zp[t], x1 = zp[t + 256], x2 = zp[t + 512];
    red[t] = x0 + x1 + x2;
    __syncthreads();
    for (int s = 128; s > 0; s >>= 1) { if (t < s) red[t] += red[t+s]; __syncthreads(); }
    float mean = red[0] * (1.0f / DD);
    __syncthreads();

    float d0 = x0 - mean, d1 = x1 - mean, d2 = x2 - mean;
    red[t] = d0*d0 + d1*d1 + d2*d2;
    __syncthreads();
    for (int s = 128; s > 0; s >>= 1) { if (t < s) red[t] += red[t+s]; __syncthreads(); }
    float rstd = rsqrtf(red[0] * (1.0f / DD) + 1e-5f);

    float y0 = d0 * rstd * g[t]       + beta[t];
    float y1 = d1 * rstd * g[t + 256] + beta[t + 256];
    float y2 = d2 * rstd * g[t + 512] + beta[t + 512];
    float* op = out + (size_t)row * DD;
    op[t] = y0; op[t + 256] = y1; op[t + 512] = y2;
    if (outh) {
        __bf16* oh = outh + (size_t)row * DD;
        oh[t] = (__bf16)y0; oh[t + 256] = (__bf16)y1; oh[t + 512] = (__bf16)y2;
    }
}

// =====================================================================
// Host orchestration
// =====================================================================
extern "C" void kernel_launch(void* const* d_in, const int* in_sizes, int n_in,
                              void* d_out, int out_size, void* d_ws, size_t ws_size,
                              hipStream_t stream) {
    const int*   tokens   = (const int*)  d_in[0];
    const int*   segments = (const int*)  d_in[1];
    const int*   vlens    = (const int*)  d_in[2];
    const float* tok_emb  = (const float*)d_in[3];
    const float* seg_emb  = (const float*)d_in[4];
    const float* pos_emb  = (const float*)d_in[5];
    const float* Wq = (const float*)d_in[6];  const float* bq = (const float*)d_in[7];
    const float* Wk = (const float*)d_in[8];  const float* bk = (const float*)d_in[9];
    const float* Wv = (const float*)d_in[10]; const float* bv = (const float*)d_in[11];
    const float* Wo = (const float*)d_in[12]; const float* bo = (const float*)d_in[13];
    const float* W1 = (const float*)d_in[14]; const float* b1 = (const float*)d_in[15];
    const float* W2 = (const float*)d_in[16]; const float* b2 = (const float*)d_in[17];
    const float* g1 = (const float*)d_in[18]; const float* be1 = (const float*)d_in[19];
    const float* g2 = (const float*)d_in[20]; const float* be2 = (const float*)d_in[21];
    float* Out = (float*)d_out;

    // ---------------- workspace carving ----------------
    const size_t nXD = (size_t)MROWS * DD;        // 6,291,456
    const size_t nP  = (size_t)BB * HH * SS * SS; // 50,331,648
    const size_t nY1 = (size_t)MROWS * FF;        // 25,165,824
    const size_t nWsm = (size_t)LL * DD * DD;     // 7,077,888
    const size_t nWff = (size_t)LL * DD * FF;     // 28,311,552

    char* wp = (char*)d_ws;
    float* X   = (float*)wp;  wp += nXD * sizeof(float);
    float* Z   = (float*)wp;  wp += nXD * sizeof(float);
    float* P   = (float*)wp;  wp += nP  * sizeof(float);
    __bf16* Xh  = (__bf16*)wp; wp += nXD * sizeof(__bf16);
    __bf16* Qh  = (__bf16*)wp; wp += nXD * sizeof(__bf16);
    __bf16* Kh  = (__bf16*)wp; wp += nXD * sizeof(__bf16);
    __bf16* Vth = (__bf16*)wp; wp += nXD * sizeof(__bf16);   // [b,h,dh,s]
    __bf16* Oh  = (__bf16*)wp; wp += nXD * sizeof(__bf16);
    __bf16* Ph  = (__bf16*)wp; wp += nP  * sizeof(__bf16);
    __bf16* Y1h = (__bf16*)wp; wp += nY1 * sizeof(__bf16);
    __bf16* WqT = (__bf16*)wp; wp += nWsm * sizeof(__bf16);  // [N=D][K=D] per layer
    __bf16* WkT = (__bf16*)wp; wp += nWsm * sizeof(__bf16);
    __bf16* WvT = (__bf16*)wp; wp += nWsm * sizeof(__bf16);
    __bf16* WoT = (__bf16*)wp; wp += nWsm * sizeof(__bf16);
    __bf16* W1T = (__bf16*)wp; wp += nWff * sizeof(__bf16);  // [N=F][K=D]
    __bf16* W2T = (__bf16*)wp; wp += nWff * sizeof(__bf16);  // [N=D][K=F]

    dim3 blk(256);

    // ---------------- weight convert/transpose (per call, deterministic) ----------------
    {
        unsigned gsm = (unsigned)((DD*(size_t)DD + 255) / 256);
        unsigned gff = (unsigned)((DD*(size_t)FF + 255) / 256);
        for (int l = 0; l < LL; ++l) {
            w_to_bf16t<<<gsm, blk, 0, stream>>>(Wq + (size_t)l*DD*DD, WqT + (size_t)l*DD*DD, DD, DD);
            w_to_bf16t<<<gsm, blk, 0, stream>>>(Wk + (size_t)l*DD*DD, WkT + (size_t)l*DD*DD, DD, DD);
            w_to_bf16t<<<gsm, blk, 0, stream>>>(Wv + (size_t)l*DD*DD, WvT + (size_t)l*DD*DD, DD, DD);
            w_to_bf16t<<<gsm, blk, 0, stream>>>(Wo + (size_t)l*DD*DD, WoT + (size_t)l*DD*DD, DD, DD);
            w_to_bf16t<<<gff, blk, 0, stream>>>(W1 + (size_t)l*DD*FF, W1T + (size_t)l*DD*FF, DD, FF);
            w_to_bf16t<<<gff, blk, 0, stream>>>(W2 + (size_t)l*DD*FF, W2T + (size_t)l*DD*FF, FF, DD);
        }
    }

    // ---------------- embeddings ----------------
    embed_kernel<<<(unsigned)((nXD + 255) / 256), blk, 0, stream>>>(
        tokens, segments, tok_emb, seg_emb, pos_emb, X, Xh);

    dim3 gLin(DD / 128, MROWS / BM, 1);      // 6  x 64
    dim3 gFF1(FF / 128, MROWS / BM, 1);      // 24 x 64
    dim3 gQK (SS / 128, SS / BM, BB * HH);   // 4 x 4 x 192
    dim3 gPV (DH / 64,  SS / BM, BB * HH);   // 1 x 4 x 192

    for (int l = 0; l < LL; ++l) {
        const __bf16* wqT = WqT + (size_t)l*DD*DD; const float* bql = bq + (size_t)l*DD;
        const __bf16* wkT = WkT + (size_t)l*DD*DD; const float* bkl = bk + (size_t)l*DD;
        const __bf16* wvT = WvT + (size_t)l*DD*DD; const float* bvl = bv + (size_t)l*DD;
        const __bf16* woT = WoT + (size_t)l*DD*DD; const float* bol = bo + (size_t)l*DD;
        const __bf16* w1T = W1T + (size_t)l*DD*FF; const float* b1l = b1 + (size_t)l*FF;
        const __bf16* w2T = W2T + (size_t)l*DD*FF; const float* b2l = b2 + (size_t)l*DD;
        const float* g1l = g1 + (size_t)l*DD; const float* be1l = be1 + (size_t)l*DD;
        const float* g2l = g2 + (size_t)l*DD; const float* be2l = be2 + (size_t)l*DD;

        // Q/K/V projections (bf16 outputs only; V stored transposed [b,h,dh,s])
        gemm_wmma<0><<<gLin, blk, 0, stream>>>(Xh, wqT, bql, nullptr, nullptr, Qh,
                                               DD, DD, DD, DD, 0, 1, 1.0f);
        gemm_wmma<0><<<gLin, blk, 0, stream>>>(Xh, wkT, bkl, nullptr, nullptr, Kh,
                                               DD, DD, DD, DD, 0, 1, 1.0f);
        gemm_wmma<0><<<gLin, blk, 0, stream>>>(Xh, wvT, bvl, nullptr, nullptr, Vth,
                                               DD, DD, DD, DD, 0, 2, 1.0f);

        // scores (fp32) -> masked softmax -> probs (bf16) -> P@V (bf16 Oh)
        gemm_wmma<1><<<gQK, blk, 0, stream>>>(Qh, Kh, nullptr, nullptr, P, nullptr,
                                              DH, 0, DD, DD, 0, 0, SCALE_QK);
        softmax_mask_kernel<<<BB * HH * SS, blk, 0, stream>>>(P, vlens, Ph);
        gemm_wmma<2><<<gPV, blk, 0, stream>>>(Ph, Vth, nullptr, nullptr, nullptr, Oh,
                                              SS, 0, SS, SS, 0, 0, 1.0f);

        // output projection + residual -> Z, LN1 -> X/Xh
        gemm_wmma<0><<<gLin, blk, 0, stream>>>(Oh, woT, bol, X, Z, nullptr,
                                               DD, DD, DD, DD, 0, 0, 1.0f);
        layernorm_kernel<<<MROWS, blk, 0, stream>>>(Z, g1l, be1l, X, Xh);

        // FFN: relu(X@W1+b1) (bf16 only) ; @W2+b2 + residual -> Z
        gemm_wmma<0><<<gFF1, blk, 0, stream>>>(Xh, w1T, b1l, nullptr, nullptr, Y1h,
                                               DD, FF, DD, DD, 1, 1, 1.0f);
        gemm_wmma<0><<<gLin, blk, 0, stream>>>(Y1h, w2T, b2l, X, Z, nullptr,
                                               FF, DD, FF, FF, 0, 0, 1.0f);

        // LN2 -> X/Xh (or final fp32 output)
        float*  dstF = (l == LL - 1) ? Out : X;
        __bf16* dstH = (l == LL - 1) ? nullptr : Xh;
        layernorm_kernel<<<MROWS, blk, 0, stream>>>(Z, g2l, be2l, dstF, dstH);
    }
}
